// BasicRecurrentLayer_56212531970292
// MI455X (gfx1250) — compile-verified, matrix operand order checked by reading, not run on hardware
//
#include <hip/hip_runtime.h>
#include <hip/hip_bf16.h>
#include <math.h>

typedef __attribute__((ext_vector_type(16))) _Float16 v16h;
typedef __attribute__((ext_vector_type(8)))  _Float16 v8h;
typedef __attribute__((ext_vector_type(4)))  _Float16 v4h;
typedef __attribute__((ext_vector_type(8)))  float    v8f;
typedef __attribute__((ext_vector_type(4)))  float    v4f;

#define B_ 64
#define T_ 1024
#define F_ 128
#define U_ 256

__device__ __forceinline__ float fast_tanh(float x) {
#if __has_builtin(__builtin_amdgcn_tanhf)
    return __builtin_amdgcn_tanhf(x);   // gfx1250 v_tanh_f32 (trans op)
#else
    return tanhf(x);
#endif
}

// Fused recurrent layer. 4 blocks of 512 threads (16 waves). Each block owns
// 16 batch rows; each wave owns a 16-column slice of U. W and R stay resident
// in VGPRs in WMMA B-operand layout (96 VGPRs/wave, under the v256 window).
// State and the input tile ping-pong through LDS in f16.
__global__ __launch_bounds__(512, 1)
void rnn_fused_kernel(const float* __restrict__ inp,   // [B,T,F]
                      const float* __restrict__ Rm,    // [F,U]
                      const float* __restrict__ Wm,    // [U,U]
                      const float* __restrict__ bias,  // [U]
                      const float* __restrict__ x0,    // [U]
                      float* __restrict__ out)         // [T,B,U]
{
    __shared__ _Float16 sstate[2][16][U_];   // 16 KB double-buffered state
    __shared__ _Float16 sinp[2][16][F_];     //  8 KB double-buffered input tile

    const int tid   = threadIdx.x;
    const int lane  = tid & 31;
    const int wave  = tid >> 5;          // 0..15
    const int l16   = lane & 15;         // A: row m ; B/C/D: column n (in-tile)
    const int khalf = lane >> 4;         // which K-half this lane holds
    const int b0    = blockIdx.x * 16;

    const int n0 = wave * 16;            // this wave's output-column slice
    const int nA = n0 + l16;             // this lane's output column

    // ---- W tiles resident in VGPRs, WMMA B-operand layout (f32 -> f16) ----
    // Element (k, n): lane%16 = n; VGPR v0..3 hold K = khalf*8 + 0..7 (pairs),
    // v4..7 hold K = 16 + khalf*8 + 0..7, per 32-wide K tile.
    v16h wreg[8];
    #pragma unroll
    for (int kt = 0; kt < 8; ++kt) {
        const int kb = kt * 32 + khalf * 8;
        v16h bt;
        #pragma unroll
        for (int j = 0; j < 8; ++j) {
            bt[j]     = (_Float16)Wm[(kb + j)      * U_ + nA];
            bt[8 + j] = (_Float16)Wm[(kb + 16 + j) * U_ + nA];
        }
        wreg[kt] = bt;
    }
    // ---- R tiles resident in VGPRs (K = F = 128 -> 4 tiles) ----
    v16h rreg[4];
    #pragma unroll
    for (int kt = 0; kt < 4; ++kt) {
        const int kb = kt * 32 + khalf * 8;
        v16h bt;
        #pragma unroll
        for (int j = 0; j < 8; ++j) {
            bt[j]     = (_Float16)Rm[(kb + j)      * U_ + nA];
            bt[8 + j] = (_Float16)Rm[(kb + 16 + j) * U_ + nA];
        }
        rreg[kt] = bt;
    }

    const float bias0 = bias[nA];

    // ---- init state buffer 0 from x0 (broadcast across the 16 batch rows) ----
    {
        const _Float16 xv = (_Float16)x0[tid & 255];
        const int m0 = (tid >> 8) * 8;           // threads 0-255: rows 0-7, 256-511: rows 8-15
        #pragma unroll
        for (int m = 0; m < 8; ++m) sstate[0][m0 + m][tid & 255] = xv;
    }
    // ---- preload input tile for t = 0 ----
    {
        const int row = wave;                    // one row per wave
        const int f4  = (lane & 31) * 4;
        const float* src = inp + ((size_t)(b0 + row) * T_ + 0) * F_ + f4;
        v4f v = *(const v4f*)src;
        v4h h; h[0] = (_Float16)v[0]; h[1] = (_Float16)v[1];
               h[2] = (_Float16)v[2]; h[3] = (_Float16)v[3];
        *(v4h*)&sinp[0][row][f4] = h;            // ds_store_b64
    }
    __syncthreads();

    const int mA = l16;                          // A-operand row this lane feeds

    for (int t = 0; t < T_; ++t) {
        const int cur = t & 1;
        const int nxt = cur ^ 1;

        v8f acc;
        #pragma unroll
        for (int r = 0; r < 8; ++r) acc[r] = bias0;

        // ---- input projection from LDS-staged f16 tile (K = 128) ----
        const _Float16* irow = &sinp[cur][mA][0];
        #pragma unroll
        for (int kt = 0; kt < 4; ++kt) {
            const int k0 = kt * 32 + khalf * 8;
            v8h lo = *(const v8h*)(irow + k0);
            v8h hi = *(const v8h*)(irow + k0 + 16);
            v16h a = __builtin_shufflevector(lo, hi, 0, 1, 2, 3, 4, 5, 6, 7,
                                             8, 9, 10, 11, 12, 13, 14, 15);
            acc = __builtin_amdgcn_wmma_f32_16x16x32_f16(false, a, false, rreg[kt],
                                                         (short)0, acc, false, false);
        }

        // ---- recurrence: state @ W (K = 256 -> 8 tiles) ----
        const _Float16* srow = &sstate[cur][mA][0];
        #pragma unroll
        for (int kt = 0; kt < 8; ++kt) {
            const int k0 = kt * 32 + khalf * 8;
            v8h lo = *(const v8h*)(srow + k0);        // ds_load_b128
            v8h hi = *(const v8h*)(srow + k0 + 16);   // ds_load_b128
            v16h a = __builtin_shufflevector(lo, hi, 0, 1, 2, 3, 4, 5, 6, 7,
                                             8, 9, 10, 11, 12, 13, 14, 15);
            acc = __builtin_amdgcn_wmma_f32_16x16x32_f16(false, a, false, wreg[kt],
                                                         (short)0, acc, false, false);
        }

        // ---- cooperative stage of next timestep's input tile ----
        if (t + 1 < T_) {
            const int row = wave;
            const int f4  = (lane & 31) * 4;
            const float* src = inp + ((size_t)(b0 + row) * T_ + (t + 1)) * F_ + f4;
            v4f v = *(const v4f*)src;
            v4h h; h[0] = (_Float16)v[0]; h[1] = (_Float16)v[1];
                   h[2] = (_Float16)v[2]; h[3] = (_Float16)v[3];
            *(v4h*)&sinp[nxt][row][f4] = h;
        }

        // ---- tanh, write f32 output, write f16 next-state to LDS ----
        float* orow = out + ((size_t)t * B_ + b0) * U_;
        #pragma unroll
        for (int r = 0; r < 8; ++r) {
            const int m = r + khalf * 8;   // C/D layout: lanes 16-31 hold M+8
            const float y = fast_tanh(acc[r]);
            orow[(size_t)m * U_ + nA] = y;
            sstate[nxt][m][nA] = (_Float16)y;
        }
        __syncthreads();
    }
}

extern "C" void kernel_launch(void* const* d_in, const int* in_sizes, int n_in,
                              void* d_out, int out_size, void* d_ws, size_t ws_size,
                              hipStream_t stream) {
    (void)in_sizes; (void)n_in; (void)out_size; (void)d_ws; (void)ws_size;
    const float* inp  = (const float*)d_in[0];
    const float* Rm   = (const float*)d_in[1];
    const float* Wm   = (const float*)d_in[2];
    const float* bias = (const float*)d_in[3];
    const float* x0   = (const float*)d_in[4];
    float* out = (float*)d_out;

    hipLaunchKernelGGL(rnn_fused_kernel, dim3(B_ / 16), dim3(512), 0, stream,
                       inp, Rm, Wm, bias, x0, out);
}
